// MagiTransformerBlock_78932908966370
// MI455X (gfx1250) — matrix operand order repaired and checked by reading, not compiled, not run on hardware
//
#include <hip/hip_runtime.h>
#include <math.h>

// Problem constants (match reference)
#define B_   2
#define S_   2048
#define DIM_ 2048
#define H_   16
#define DH_  128
#define FF_  8192

typedef __attribute__((ext_vector_type(16))) __bf16 v16bf;
typedef __attribute__((ext_vector_type(8)))  __bf16 v8bf;
typedef __attribute__((ext_vector_type(4)))  __bf16 v4bf;
typedef __attribute__((ext_vector_type(2)))  __bf16 v2bf;
typedef __attribute__((ext_vector_type(8)))  float  v8f;
typedef unsigned int u32x4 __attribute__((ext_vector_type(4)));
typedef int          i32x8 __attribute__((ext_vector_type(8)));
typedef int          i32x4 __attribute__((ext_vector_type(4)));

// Tensor Data Mover path (gfx1250): guarded so host pass / older toolchains
// fall back to the proven manual global->regs->LDS pipeline.
#if defined(__AMDGCN__) && __has_builtin(__builtin_amdgcn_tensor_load_to_lds)
#define USE_TDM 1
#else
#define USE_TDM 0
#endif

#if USE_TDM
// DMA one 128x32 bf16 A-tile (row-major, row stride K elements) into LDS with
// hardware padding: 8 bf16 (4 dwords) pad after every 32 bf16 (16 dwords),
// reproducing the LD=40 LDS layout. D# bit layout per CDNA5 ISA 8.3/8.4.
__device__ __forceinline__ void tdm_load_a_tile(const __bf16* g, __bf16* lds, int K) {
    unsigned long long ga = (unsigned long long)(size_t)g;
    // Flat shared-aperture pointers carry the LDS byte address in addr[31:0].
    unsigned ldsa = (unsigned)(size_t)lds;
    u32x4 g0;
    g0[0] = 1u;                                                 // count=1, user desc
    g0[1] = ldsa;                                               // lds_addr (bytes)
    g0[2] = (unsigned)ga;                                       // global_addr[31:0]
    g0[3] = (unsigned)((ga >> 32) & 0x1FFFFFFu) | (2u << 30);   // addr[56:32]|type=2
    i32x8 g1;
    g1[0] = (1 << 16)        // data_size = 2 bytes
          | (1 << 20)        // pad_enable
          | (3 << 22)        // pad_interval: 16 dwords (64B = one 32-elem row)
          | (3 << 25);       // pad_amount: 4 dwords (16B = 8 bf16)
    g1[1] = (int)(32u << 16);    // tensor_dim0 = 32  (bits 79:48, low half)
    g1[2] = (int)(128u << 16);   // tensor_dim1 = 128 (bits 111:80, low half)
    g1[3] = (int)(32u << 16);    // tile_dim0 = 32    (bits 127:112)
    g1[4] = 128;                 // tile_dim1 = 128   (bits 143:128)
    g1[5] = K;                   // tensor_dim0_stride[31:0] (elements)
    g1[6] = 0;
    g1[7] = 0;
    i32x4 z4 = {0, 0, 0, 0};
#if __clang_major__ >= 23
    i32x8 z8 = {0, 0, 0, 0, 0, 0, 0, 0};
    __builtin_amdgcn_tensor_load_to_lds(g0, g1, z4, z4, z8, 0);
#else
    __builtin_amdgcn_tensor_load_to_lds(g0, g1, z4, z4, 0);
#endif
}
#endif

// ---------------------------------------------------------------------------
// WMMA fragment loaders (CDNA5 16x16x32 bf16 VGPR layouts, wave32)
// ---------------------------------------------------------------------------
__device__ __forceinline__ v16bf frag_a(const __bf16* t, int ld, int row0, int k0) {
    const int lane = threadIdx.x & 31;
    const __bf16* p = t + (row0 + (lane & 15)) * ld + k0 + ((lane >> 4) << 3);
    v8bf lo = *(const v8bf*)(p);
    v8bf hi = *(const v8bf*)(p + 16);
    v16bf r;
#pragma unroll
    for (int i = 0; i < 8; ++i) { r[i] = lo[i]; r[i + 8] = hi[i]; }
    return r;
}

__device__ __forceinline__ v16bf frag_b(const __bf16* t, int ld, int col0, int k0) {
    const int lane = threadIdx.x & 31;
    const __bf16* p = t + (col0 + (lane & 15)) * ld + k0 + ((lane >> 4) << 3);
    v8bf lo = *(const v8bf*)(p);
    v8bf hi = *(const v8bf*)(p + 16);
    v16bf r;
#pragma unroll
    for (int i = 0; i < 8; ++i) { r[i] = lo[i]; r[i + 8] = hi[i]; }
    return r;
}

#define WMMA_BF16(a, b, c) \
    __builtin_amdgcn_wmma_f32_16x16x32_bf16(false, (a), false, (b), (short)0, (c), false, false)

// ---------------------------------------------------------------------------
// fp32 -> bf16 conversion (weights), float4 -> v4bf
// ---------------------------------------------------------------------------
__global__ __launch_bounds__(256)
void cvt_kernel(const float* __restrict__ s, __bf16* __restrict__ d, int n4) {
    int i = blockIdx.x * 256 + threadIdx.x;
    if (i < n4) {
        float4 v = ((const float4*)s)[i];
        v4bf t;
        t[0] = (__bf16)v.x; t[1] = (__bf16)v.y; t[2] = (__bf16)v.z; t[3] = (__bf16)v.w;
        ((v4bf*)d)[i] = t;
    }
}

// ---------------------------------------------------------------------------
// Row LayerNorm over D elements, fp32 in -> bf16 out. One block per row.
// ---------------------------------------------------------------------------
__global__ __launch_bounds__(256)
void ln_kernel(const float* __restrict__ x, const float* __restrict__ w,
               const float* __restrict__ b, __bf16* __restrict__ y, int D) {
    __shared__ float s1[8], s2[8];
    const int row = blockIdx.x;
    const float* xr = x + (size_t)row * D;
    float a1 = 0.f, a2 = 0.f;
    for (int i = threadIdx.x; i < D; i += 256) { float v = xr[i]; a1 += v; a2 += v * v; }
#pragma unroll
    for (int off = 16; off; off >>= 1) {
        a1 += __shfl_xor(a1, off, 32);
        a2 += __shfl_xor(a2, off, 32);
    }
    if ((threadIdx.x & 31) == 0) { s1[threadIdx.x >> 5] = a1; s2[threadIdx.x >> 5] = a2; }
    __syncthreads();
    a1 = 0.f; a2 = 0.f;
#pragma unroll
    for (int i = 0; i < 8; ++i) { a1 += s1[i]; a2 += s2[i]; }
    const float mean = a1 / (float)D;
    const float var  = a2 / (float)D - mean * mean;
    const float rs   = rsqrtf(var + 1e-5f);
    __bf16* yr = y + (size_t)row * D;
    for (int i = threadIdx.x; i < D; i += 256)
        yr[i] = (__bf16)((xr[i] - mean) * rs * w[i] + b[i]);
}

// ---------------------------------------------------------------------------
// Tiled bf16 WMMA GEMM, double-buffered. A,B bf16 in HBM; C fp32 or bf16.
// A-tiles: TDM async DMA to LDS (padded) when available, else manual copy.
// B-tiles: manual transpose to [n][k] via v8bf loads + paired v2bf stores.
// 128x128 block, BK=32, 256 threads = 8 waves, wave = 32x64 (2x4 accs).
// ---------------------------------------------------------------------------
template <int GELU, int RES, int OUTBF>
__global__ __launch_bounds__(256)
void gemm_kernel(const __bf16* __restrict__ A, const __bf16* __restrict__ Bw,
                 const float* __restrict__ bias, const float* __restrict__ res,
                 void* __restrict__ Cv, int M, int N, int K) {
    constexpr int LD = 40;                       // 32 + 8 pad (80B rows, 16B aligned)
    __shared__ __bf16 As[2][128 * LD];
    __shared__ __bf16 Bs[2][128 * LD];           // stored [n][k]
    const int tid  = threadIdx.x;
    const int wid  = tid >> 5;
    const int wr   = wid >> 1, wc = wid & 1;     // 4x2 wave grid
    const int lane = tid & 31, hf = lane >> 4, l16 = lane & 15;
    const int m0 = blockIdx.y * 128, n0 = blockIdx.x * 128;

    // B tile 32x128: 256 slots (2 k-rows x 8 n-cols), exactly 1 per thread.
    const int bkr = (tid >> 4) << 1, bc = (tid & 15) << 3;
    auto loadB = [&](int k0, v8bf* rb) {
        rb[0] = *(const v8bf*)(Bw + (size_t)(k0 + bkr + 0) * N + n0 + bc);
        rb[1] = *(const v8bf*)(Bw + (size_t)(k0 + bkr + 1) * N + n0 + bc);
    };
    auto storeB = [&](int buf, const v8bf* rb) {
#pragma unroll
        for (int e = 0; e < 8; ++e) {
            v2bf t; t[0] = rb[0][e]; t[1] = rb[1][e];
            *(v2bf*)&Bs[buf][(bc + e) * LD + bkr] = t;
        }
    };
#if !USE_TDM
    auto loadA = [&](int k0, v8bf* ra) {
#pragma unroll
        for (int it = 0; it < 2; ++it) {
            int s = tid + it * 256;              // 512 v8bf chunks
            int r = s >> 2, c = (s & 3) << 3;
            ra[it] = *(const v8bf*)(A + (size_t)(m0 + r) * K + k0 + c);
        }
    };
    auto storeA = [&](int buf, const v8bf* ra) {
#pragma unroll
        for (int it = 0; it < 2; ++it) {
            int s = tid + it * 256;
            int r = s >> 2, c = (s & 3) << 3;
            *(v8bf*)&As[buf][r * LD + c] = ra[it];
        }
    };
#endif

    v8f acc[2][4];
#pragma unroll
    for (int i = 0; i < 2; ++i)
#pragma unroll
        for (int j = 0; j < 4; ++j)
#pragma unroll
            for (int e = 0; e < 8; ++e) acc[i][j][e] = 0.f;

    v8bf rb[2];
#if USE_TDM
    if (wid == 0) tdm_load_a_tile(A + (size_t)m0 * K, &As[0][0], K);
    loadB(0, rb); storeB(0, rb);
    if (K > 32) loadB(32, rb);
    if (wid == 0) __builtin_amdgcn_s_wait_tensorcnt(0);
    __syncthreads();
#else
    v8bf ra[2];
    loadA(0, ra); loadB(0, rb);
    storeA(0, ra); storeB(0, rb);
    if (K > 32) { loadA(32, ra); loadB(32, rb); }
    __syncthreads();
#endif

    int cur = 0;
    for (int k0 = 0; k0 < K; k0 += 32) {
        const int nxt = cur ^ 1;
        if (k0 + 32 < K) {
            storeB(nxt, rb);                     // waits on iter-old B loads
            if (k0 + 64 < K) loadB(k0 + 64, rb);
#if USE_TDM
            if (wid == 0)
                tdm_load_a_tile(A + (size_t)m0 * K + (k0 + 32), &As[nxt][0], K);
#else
            storeA(nxt, ra);
            if (k0 + 64 < K) loadA(k0 + 64, ra);
#endif
        }
        // batch all fragment loads, then stream 8 WMMAs
        v16bf a0 = frag_a(As[cur], LD, wr * 32 + 0, 0);
        v16bf a1 = frag_a(As[cur], LD, wr * 32 + 16, 0);
        v16bf bf[4];
#pragma unroll
        for (int j = 0; j < 4; ++j) bf[j] = frag_b(Bs[cur], LD, wc * 64 + j * 16, 0);
#pragma unroll
        for (int j = 0; j < 4; ++j) {
            acc[0][j] = WMMA_BF16(a0, bf[j], acc[0][j]);
            acc[1][j] = WMMA_BF16(a1, bf[j], acc[1][j]);
        }
#if USE_TDM
        if (k0 + 32 < K && wid == 0) __builtin_amdgcn_s_wait_tensorcnt(0);
#endif
        __syncthreads();
        cur = nxt;
    }

    // epilogue: bias -> gelu -> residual -> (fp32|bf16) store
#pragma unroll
    for (int i = 0; i < 2; ++i)
#pragma unroll
        for (int j = 0; j < 4; ++j)
#pragma unroll
            for (int r = 0; r < 8; ++r) {
                int row = m0 + wr * 32 + i * 16 + r + hf * 8;
                int col = n0 + wc * 64 + j * 16 + l16;
                float v = acc[i][j][r];
                if (bias) v += bias[col];
                if (GELU) v = 0.5f * v * (1.f + erff(v * 0.70710678118654752f));
                if (RES)  v += res[(size_t)row * N + col];
                size_t idx = (size_t)row * N + col;
                if (OUTBF) ((__bf16*)Cv)[idx] = (__bf16)v;
                else       ((float*)Cv)[idx]  = v;
            }
}

// ---------------------------------------------------------------------------
// Per-head LayerNorm(DH) + RoPE, transpose [B,S,H,DH] -> [B*H,S,DH]. bf16 io.
// doNR=0: pure transpose (V path). One 128-thread block per (b,s,h).
// ---------------------------------------------------------------------------
__global__ __launch_bounds__(128)
void headop_kernel(const __bf16* __restrict__ in, const float* __restrict__ w,
                   const float* __restrict__ b, const float* __restrict__ fcos,
                   const float* __restrict__ fsin, __bf16* __restrict__ out, int doNR) {
    __shared__ float shm[DH_];
    __shared__ float r1[4], r2[4];
    const int bid = blockIdx.x;
    const int h  = bid % H_;
    const int s  = (bid / H_) % S_;
    const int bb = bid / (H_ * S_);
    const int d  = threadIdx.x;
    float v = (float)in[(((size_t)bb * S_ + s) * H_ + h) * DH_ + d];
    __bf16* op = out + (((size_t)bb * H_ + h) * S_ + s) * DH_;
    if (!doNR) { op[d] = (__bf16)v; return; }
    float a1 = v, a2 = v * v;
#pragma unroll
    for (int off = 16; off; off >>= 1) {
        a1 += __shfl_xor(a1, off, 32);
        a2 += __shfl_xor(a2, off, 32);
    }
    if ((d & 31) == 0) { r1[d >> 5] = a1; r2[d >> 5] = a2; }
    __syncthreads();
    a1 = r1[0] + r1[1] + r1[2] + r1[3];
    a2 = r2[0] + r2[1] + r2[2] + r2[3];
    const float mean = a1 / (float)DH_;
    const float rs   = rsqrtf(a2 / (float)DH_ - mean * mean + 1e-5f);
    shm[d] = (v - mean) * rs * w[d] + b[d];
    __syncthreads();
    if (d < DH_ / 2) {
        float x0 = shm[2 * d], x1 = shm[2 * d + 1];
        float c  = fcos[(size_t)s * (DH_ / 2) + d];
        float sn = fsin[(size_t)s * (DH_ / 2) + d];
        op[2 * d]     = (__bf16)(x0 * c - x1 * sn);
        op[2 * d + 1] = (__bf16)(x0 * sn + x1 * c);
    }
}

// ---------------------------------------------------------------------------
// Flash attention, non-causal. Q/K/V bf16 [B*H,S,DH] -> Out bf16 [B,S,H*DH].
// BQ=BKV=64, 128 threads = 4 waves; bf16 tiles are raw v8bf copies (no cvt).
// ---------------------------------------------------------------------------
__global__ __launch_bounds__(128)
void attn_kernel(const __bf16* __restrict__ Q, const __bf16* __restrict__ Kt,
                 const __bf16* __restrict__ V, __bf16* __restrict__ Out) {
    constexpr int LQK = 136;   // DH + 8
    constexpr int LV  = 72;    // 64 + 8
    constexpr int LP  = 72;    // 64 + 8
    __shared__ __bf16 Qs[64 * LQK];
    __shared__ __bf16 Ks[64 * LQK];   // [key][d]  == B-layout [n=key][k=d]
    __shared__ __bf16 Vs[128 * LV];   // [d][key]  == B-layout [n=d][k=key]
    __shared__ __bf16 Ps[64 * LP];    // [q][key]  == A-layout

    const int tid  = threadIdx.x;
    const int wv_  = tid >> 5;                 // wave 0..3
    const int lane = tid & 31, hf = lane >> 4, l16 = lane & 15;
    const int bh = blockIdx.y;
    const int bb = bh / H_, h = bh % H_;
    const int q0 = blockIdx.x * 64;
    const float scale = 0.088388347648318447f; // 1/sqrt(DH)

    // stage Q tile: 1024 v8bf chunks, straight copy
    const __bf16* Qp = Q + ((size_t)bh * S_ + q0) * DH_;
#pragma unroll
    for (int it = 0; it < 8; ++it) {
        int s = tid + it * 128;
        int r = s >> 4, c = (s & 15) << 3;
        *(v8bf*)&Qs[r * LQK + c] = *(const v8bf*)(Qp + (size_t)r * DH_ + c);
    }

    v8f   o[8];
    float mrow[8], lrow[8];
#pragma unroll
    for (int j = 0; j < 8; ++j) {
        mrow[j] = -3.0e38f; lrow[j] = 0.f;
#pragma unroll
        for (int e = 0; e < 8; ++e) o[j][e] = 0.f;
    }

    for (int kv0 = 0; kv0 < S_; kv0 += 64) {
        __syncthreads();                       // previous iter done with Ks/Vs
        const __bf16* Kp = Kt + ((size_t)bh * S_ + kv0) * DH_;
        const __bf16* Vp = V  + ((size_t)bh * S_ + kv0) * DH_;
#pragma unroll
        for (int it = 0; it < 8; ++it) {       // K tile straight copy
            int s = tid + it * 128;
            int r = s >> 4, c = (s & 15) << 3;
            *(v8bf*)&Ks[r * LQK + c] = *(const v8bf*)(Kp + (size_t)r * DH_ + c);
        }
#pragma unroll
        for (int it = 0; it < 4; ++it) {       // V tile transposed, key-pairs
            int s = tid + it * 128;            // 512 slots
            int kr = (s >> 4) << 1, c = (s & 15) << 3;
            v8bf v0 = *(const v8bf*)(Vp + (size_t)(kr + 0) * DH_ + c);
            v8bf v1 = *(const v8bf*)(Vp + (size_t)(kr + 1) * DH_ + c);
#pragma unroll
            for (int e = 0; e < 8; ++e) {
                v2bf t; t[0] = v0[e]; t[1] = v1[e];
                *(v2bf*)&Vs[(c + e) * LV + kr] = t;
            }
        }
        __syncthreads();

        // S = Q @ K^T (16x64 per wave), k over DH in 4 steps
        v8f sacc[4];
#pragma unroll
        for (int j = 0; j < 4; ++j)
#pragma unroll
            for (int e = 0; e < 8; ++e) sacc[j][e] = 0.f;
#pragma unroll
        for (int kk = 0; kk < DH_; kk += 32) {
            v16bf a = frag_a(Qs, LQK, wv_ * 16, kk);
            v16bf bq[4];
#pragma unroll
            for (int j = 0; j < 4; ++j) bq[j] = frag_b(Ks, LQK, j * 16, kk);
#pragma unroll
            for (int j = 0; j < 4; ++j) sacc[j] = WMMA_BF16(a, bq[j], sacc[j]);
        }

        // online softmax: acc row r lives in vgpr r across a 16-lane half
#pragma unroll
        for (int r = 0; r < 8; ++r) {
            float rm = -3.0e38f;
#pragma unroll
            for (int j = 0; j < 4; ++j) {
                float v = sacc[j][r] * scale;
                sacc[j][r] = v;
                rm = fmaxf(rm, v);
            }
#pragma unroll
            for (int off = 1; off < 16; off <<= 1) rm = fmaxf(rm, __shfl_xor(rm, off, 32));
            const float mn    = fmaxf(mrow[r], rm);
            const float alpha = __expf(mrow[r] - mn);
            float rsum = 0.f;
#pragma unroll
            for (int j = 0; j < 4; ++j) {
                float p = __expf(sacc[j][r] - mn);
                sacc[j][r] = p;
                rsum += p;
            }
#pragma unroll
            for (int off = 1; off < 16; off <<= 1) rsum += __shfl_xor(rsum, off, 32);
            lrow[r] = lrow[r] * alpha + rsum;
            mrow[r] = mn;
#pragma unroll
            for (int j2 = 0; j2 < 8; ++j2) o[j2][r] *= alpha;
            const int prow = wv_ * 16 + r + hf * 8;
#pragma unroll
            for (int j = 0; j < 4; ++j)
                Ps[prow * LP + j * 16 + l16] = (__bf16)sacc[j][r];
        }

        // O += P @ V (k over 64 keys in 2 steps): batch frags, stream WMMAs
#pragma unroll
        for (int kk = 0; kk < 64; kk += 32) {
            v16bf a = frag_a(Ps, LP, wv_ * 16, kk);
            v16bf bv[8];
#pragma unroll
            for (int j = 0; j < 8; ++j) bv[j] = frag_b(Vs, LV, j * 16, kk);
#pragma unroll
            for (int j = 0; j < 8; ++j) o[j] = WMMA_BF16(a, bv[j], o[j]);
        }
    }

    // normalize and write bf16 [B,S,H*DH]
#pragma unroll
    for (int j = 0; j < 8; ++j)
#pragma unroll
        for (int r = 0; r < 8; ++r) {
            int row = q0 + wv_ * 16 + r + hf * 8;
            int col = j * 16 + l16;
            Out[((size_t)bb * S_ + row) * (H_ * DH_) + h * DH_ + col] =
                (__bf16)(o[j][r] / lrow[r]);
        }
}

// ---------------------------------------------------------------------------
// Launcher
// ---------------------------------------------------------------------------
extern "C" void kernel_launch(void* const* d_in, const int* in_sizes, int n_in,
                              void* d_out, int out_size, void* d_ws, size_t ws_size,
                              hipStream_t stream) {
    (void)in_sizes; (void)n_in; (void)out_size; (void)ws_size;
    const float* x    = (const float*)d_in[0];
    const float* fcos = (const float*)d_in[1];
    const float* fsin = (const float*)d_in[2];
    const float* wq   = (const float*)d_in[3];
    const float* wk   = (const float*)d_in[4];
    const float* wvp  = (const float*)d_in[5];
    const float* nqw  = (const float*)d_in[6];
    const float* nqb  = (const float*)d_in[7];
    const float* nkw  = (const float*)d_in[8];
    const float* nkb  = (const float*)d_in[9];
    const float* wo   = (const float*)d_in[10];
    const float* bo   = (const float*)d_in[11];
    const float* ln1w = (const float*)d_in[12];
    const float* ln1b = (const float*)d_in[13];
    const float* ln3w = (const float*)d_in[14];
    const float* ln3b = (const float*)d_in[15];
    const float* w1   = (const float*)d_in[16];
    const float* b1   = (const float*)d_in[17];
    const float* w2   = (const float*)d_in[18];
    const float* b2   = (const float*)d_in[19];

    const int M = B_ * S_;                       // 4096 rows
    const size_t HS = (size_t)M * DIM_;          // 8M elements per activation slab
    __bf16* wsb = (__bf16*)d_ws;
    __bf16* n1h = wsb + 0 * HS;
    __bf16* qh  = wsb + 1 * HS;
    __bf16* kh  = wsb + 2 * HS;
    __bf16* vh  = wsb + 3 * HS;
    __bf16* qrh = wsb + 4 * HS;
    __bf16* krh = wsb + 5 * HS;
    __bf16* vrh = wsb + 6 * HS;
    __bf16* attnh = qh;                          // q consumed before attention
    __bf16* n3h   = n1h;                         // n1 consumed by QKV GEMMs
    __bf16* ffh   = vh;                          // slabs 3..6 (M*FF bf16), consumed
    float*  x1    = (float*)(wsb + 7 * HS);      // 8M floats (2 slabs of bf16 units)
    __bf16* wqh = wsb + 9 * HS;
    __bf16* wkh = wqh + (size_t)DIM_ * DIM_;
    __bf16* wvh = wkh + (size_t)DIM_ * DIM_;
    __bf16* woh = wvh + (size_t)DIM_ * DIM_;
    __bf16* w1h = woh + (size_t)DIM_ * DIM_;
    __bf16* w2h = w1h + (size_t)DIM_ * FF_;
    float*  out = (float*)d_out;

    // 0) weights fp32 -> bf16 (once per launch, deterministic)
    auto cvt = [&](const float* s, __bf16* d, size_t n) {
        int n4 = (int)(n / 4);
        cvt_kernel<<<(n4 + 255) / 256, 256, 0, stream>>>(s, d, n4);
    };
    cvt(wq, wqh, (size_t)DIM_ * DIM_);
    cvt(wk, wkh, (size_t)DIM_ * DIM_);
    cvt(wvp, wvh, (size_t)DIM_ * DIM_);
    cvt(wo, woh, (size_t)DIM_ * DIM_);
    cvt(w1, w1h, (size_t)DIM_ * FF_);
    cvt(w2, w2h, (size_t)DIM_ * FF_);

    // 1) LN1 (fp32 -> bf16)
    ln_kernel<<<M, 256, 0, stream>>>(x, ln1w, ln1b, n1h, DIM_);

    // 2) QKV projections (bf16 out)
    dim3 g1(DIM_ / 128, M / 128);
    gemm_kernel<0, 0, 1><<<g1, 256, 0, stream>>>(n1h, wqh, nullptr, nullptr, qh, M, DIM_, DIM_);
    gemm_kernel<0, 0, 1><<<g1, 256, 0, stream>>>(n1h, wkh, nullptr, nullptr, kh, M, DIM_, DIM_);
    gemm_kernel<0, 0, 1><<<g1, 256, 0, stream>>>(n1h, wvh, nullptr, nullptr, vh, M, DIM_, DIM_);

    // 3) per-head LN + RoPE (q,k), transpose (v)
    headop_kernel<<<B_ * S_ * H_, DH_, 0, stream>>>(qh, nqw, nqb, fcos, fsin, qrh, 1);
    headop_kernel<<<B_ * S_ * H_, DH_, 0, stream>>>(kh, nkw, nkb, fcos, fsin, krh, 1);
    headop_kernel<<<B_ * S_ * H_, DH_, 0, stream>>>(vh, nullptr, nullptr, nullptr, nullptr, vrh, 0);

    // 4) attention (bf16 in/out)
    dim3 ga(S_ / 64, B_ * H_);
    attn_kernel<<<ga, 128, 0, stream>>>(qrh, krh, vrh, attnh);

    // 5) output projection + bias + residual(x) -> x1 (fp32)
    gemm_kernel<0, 1, 0><<<g1, 256, 0, stream>>>(attnh, woh, bo, x, x1, M, DIM_, DIM_);

    // 6) LN3 (fp32 -> bf16)
    ln_kernel<<<M, 256, 0, stream>>>(x1, ln3w, ln3b, n3h, DIM_);

    // 7) MLP up + exact GELU (bf16 out)
    dim3 g2(FF_ / 128, M / 128);
    gemm_kernel<1, 0, 1><<<g2, 256, 0, stream>>>(n3h, w1h, b1, nullptr, ffh, M, FF_, DIM_);

    // 8) MLP down + bias + residual(x1) -> d_out (fp32)
    gemm_kernel<0, 1, 0><<<g1, 256, 0, stream>>>(ffh, w2h, b2, x1, out, M, DIM_, FF_);
}
